// LSTM_32487132627516
// MI455X (gfx1250) — matrix operand order repaired and checked by reading, not compiled, not run on hardware
//
#include <hip/hip_runtime.h>
#include <hip/hip_bf16.h>
#include <math.h>

// LSTM: B=64, T=512, I=H=1024.
// Phase 1: xg[t][b][4H] = x @ Wx + bsum  (f16 WMMA GEMM, compute-bound)
// Phase 2: 512 sequential fused step kernels; WhT (8 MB f16) is L2-resident
//          (192 MB L2). K split 4-ways per tile + LDS reduction for parallelism.
// k-loops fully unrolled with ping-pong buffers; __builtin_amdgcn_sched_barrier
// fences pin the prefetch distance so each WMMA group waits only partially
// (next stage's loads already in flight) instead of draining loadcnt to 0.

#define B_ 64
#define T_ 512
#define I_ 1024
#define H_ 1024
#define G4H 4096

#if defined(__has_builtin)
#if __has_builtin(__builtin_amdgcn_sched_barrier)
#define SCHED_FENCE() __builtin_amdgcn_sched_barrier(0)
#endif
#endif
#ifndef SCHED_FENCE
#define SCHED_FENCE()
#endif

typedef __attribute__((ext_vector_type(16))) _Float16 v16h;
typedef __attribute__((ext_vector_type(8)))  _Float16 v8h;
typedef __attribute__((ext_vector_type(8)))  float    v8f;

// ---- WMMA operand loaders (ISA 7.12.2 layouts, wave32) -----------------------

// A: 16x32 f16 (M x K), row-major, row stride lda (halves).
// lane<16: row=lane,     halves 0-7 = K 0-7,  halves 8-15 = K 16-23
// lane>=16: row=lane-16, halves 0-7 = K 8-15, halves 8-15 = K 24-31
__device__ __forceinline__ v16h load_a16x32(const _Float16* A, int lda,
                                            int m0, int k0, int lane) {
  const int row = m0 + (lane & 15);
  const int kb  = (lane >> 4) << 3;                  // 0 or 8
  const _Float16* p = A + (size_t)row * lda + k0 + kb;
  union { v16h v; v8h h[2]; } u;
  u.h[0] = *(const v8h*)(p);                         // K = kb .. kb+7
  u.h[1] = *(const v8h*)(p + 16);                    // K = kb+16 .. kb+23
  return u.v;
}

// B: 32x16 f16 (K x N) from transposed storage BT[n][k], row stride ldb.
// lane<16: col=lane, K 0-7,16-23 ; lane>=16: col=lane-16, K 8-15,24-31.
__device__ __forceinline__ v16h load_b32x16(const _Float16* BT, int ldb,
                                            int n0, int k0, int lane) {
  const int col = n0 + (lane & 15);
  const int kb  = (lane >> 4) << 3;                  // 0 or 8
  const _Float16* p = BT + (size_t)col * ldb + k0 + kb;
  union { v16h v; v8h h[2]; } u;
  u.h[0] = *(const v8h*)(p);
  u.h[1] = *(const v8h*)(p + 16);
  return u.v;
}

__device__ __forceinline__ float sigm(float x) { return 1.0f / (1.0f + __expf(-x)); }

// ---- preprocessing ----------------------------------------------------------

__global__ __launch_bounds__(256) void cvt_x_f16(const float* __restrict__ x,
                                                 _Float16* __restrict__ xf, int n) {
  int stride = gridDim.x * blockDim.x;
  for (int i = blockIdx.x * blockDim.x + threadIdx.x; i < n; i += stride)
    xf[i] = (_Float16)x[i];
}

// dst[n][k] = src_gate[k][n%1024], gates concatenated along n: {f,i,g,o}
__global__ __launch_bounds__(256) void transpose_concat(
    _Float16* __restrict__ dst, const float* __restrict__ sf,
    const float* __restrict__ si, const float* __restrict__ sg,
    const float* __restrict__ so) {
  const int total = G4H * 1024;
  int stride = gridDim.x * blockDim.x;
  for (int idx = blockIdx.x * blockDim.x + threadIdx.x; idx < total; idx += stride) {
    int n = idx >> 10, k = idx & 1023;
    int gate = n >> 10, cn = n & 1023;
    const float* s = (gate == 0) ? sf : (gate == 1) ? si : (gate == 2) ? sg : so;
    dst[idx] = (_Float16)s[(size_t)k * 1024 + cn];
  }
}

__global__ __launch_bounds__(256) void build_bsum(
    float* __restrict__ out,
    const float* bif, const float* bhf, const float* bii, const float* bhi,
    const float* big, const float* bhg, const float* bio, const float* bho) {
  int idx = blockIdx.x * blockDim.x + threadIdx.x;
  if (idx >= G4H) return;
  int gate = idx >> 10, c = idx & 1023;
  float v;
  if      (gate == 0) v = bif[c] + bhf[c];
  else if (gate == 1) v = bii[c] + bhi[c];
  else if (gate == 2) v = big[c] + bhg[c];
  else                v = bio[c] + bho[c];
  out[idx] = v;
}

__global__ __launch_bounds__(256) void init_state(float* __restrict__ c,
                                                  float* __restrict__ h32,
                                                  _Float16* __restrict__ h16) {
  int stride = gridDim.x * blockDim.x;
  for (int i = blockIdx.x * blockDim.x + threadIdx.x; i < B_ * H_; i += stride) {
    c[i] = 0.0f;
    h32[i] = 0.0f;
    h16[i] = (_Float16)0.0f;            // buffer 0 (read at t=0)
    h16[i + B_ * H_] = (_Float16)0.0f;  // buffer 1
  }
}

// ---- phase 1: xg = x @ Wx + bsum --------------------------------------------
// One wave computes a 16(M) x 64(N) tile. k-loop FULLY unrolled with ping-pong
// buffers. jobs = 2048 * 64 = 131072 waves = 16384 blocks x 8 waves.
__global__ __launch_bounds__(256, 1) void xg_gemm(const _Float16* __restrict__ xf,
                                                  const _Float16* __restrict__ WxT,
                                                  const float* __restrict__ bsum,
                                                  float* __restrict__ xg) {
  const int lane = threadIdx.x & 31;
  const int job  = blockIdx.x * 8 + (threadIdx.x >> 5);
  const int m0   = (job >> 6) << 4;
  const int n0   = (job & 63) << 6;
  const v8f zero = {0.f, 0.f, 0.f, 0.f, 0.f, 0.f, 0.f, 0.f};
  v8f acc[4] = {zero, zero, zero, zero};

  v16h a0 = load_a16x32(xf, I_, m0, 0, lane);
  v16h b0[4];
#pragma unroll
  for (int s = 0; s < 4; ++s) b0[s] = load_b32x16(WxT, I_, n0 + 16 * s, 0, lane);

#pragma unroll
  for (int k0 = 0; k0 < I_; k0 += 64) {
    // issue loads for k0+32 (buf1), then fence, then compute with buf0
    v16h a1 = load_a16x32(xf, I_, m0, k0 + 32, lane);
    v16h b1[4];
#pragma unroll
    for (int s = 0; s < 4; ++s)
      b1[s] = load_b32x16(WxT, I_, n0 + 16 * s, k0 + 32, lane);
    SCHED_FENCE();
#pragma unroll
    for (int s = 0; s < 4; ++s)
      acc[s] = __builtin_amdgcn_wmma_f32_16x16x32_f16(
          false, a0, false, b0[s], (short)0, acc[s], false, false);
    SCHED_FENCE();

    // issue loads for k0+64 (buf0; dropped at compile time on last iteration),
    // then fence, then compute with buf1
    if (k0 + 64 < I_) {
      a0 = load_a16x32(xf, I_, m0, k0 + 64, lane);
#pragma unroll
      for (int s = 0; s < 4; ++s)
        b0[s] = load_b32x16(WxT, I_, n0 + 16 * s, k0 + 64, lane);
    }
    SCHED_FENCE();
#pragma unroll
    for (int s = 0; s < 4; ++s)
      acc[s] = __builtin_amdgcn_wmma_f32_16x16x32_f16(
          false, a1, false, b1[s], (short)0, acc[s], false, false);
    SCHED_FENCE();
  }

  const int hi8 = (lane >> 4) << 3;
#pragma unroll
  for (int s = 0; s < 4; ++s) {
    const int N = n0 + 16 * s + (lane & 15);
    const float bias = bsum[N];
#pragma unroll
    for (int e = 0; e < 8; ++e) {
      int r = m0 + hi8 + e;            // flattened row = b*T + t
      int bidx = r >> 9, tidx = r & 511;
      xg[(size_t)(tidx * B_ + bidx) * G4H + N] = acc[s][e] + bias;
    }
  }
}

// ---- phase 2: one fused step ------------------------------------------------
// 128 blocks x 8 waves. Block = (batch tile mt 0..3) x (n-tile pair 0..31).
// Wave w -> nt_local = w>>2, kq = w&3 (K quarter, 8 k-steps). 4 gates per wave.
// Partials reduced through LDS; wave with kq==0 does gate math + state update.
__global__ __launch_bounds__(256, 1) void lstm_step(const float* __restrict__ xg_t,
                                                    const _Float16* __restrict__ h_in,
                                                    const _Float16* __restrict__ WhT,
                                                    _Float16* __restrict__ h_out,
                                                    float* __restrict__ c,
                                                    float* __restrict__ h32) {
  __shared__ float red[8][4][256];     // [wave][gate][lane*8+e]  (32 KB)

  const int lane = threadIdx.x & 31;
  const int w    = threadIdx.x >> 5;   // 0..7
  const int mt   = blockIdx.x >> 5;    // 0..3
  const int np   = blockIdx.x & 31;    // n-tile pair
  const int m0   = mt << 4;
  const int nt_local = w >> 2;
  const int kq   = w & 3;
  const int n0   = ((np << 1) + nt_local) << 4;   // column tile within a gate
  const int kbase = kq << 8;                      // K quarter start

  const v8f zero = {0.f, 0.f, 0.f, 0.f, 0.f, 0.f, 0.f, 0.f};
  v8f acc[4] = {zero, zero, zero, zero};

  v16h a0 = load_a16x32(h_in, H_, m0, kbase, lane);
  v16h b0[4];
#pragma unroll
  for (int g = 0; g < 4; ++g)
    b0[g] = load_b32x16(WhT, H_, (g << 10) + n0, kbase, lane);

#pragma unroll
  for (int ki = 0; ki < 8; ki += 2) {
    v16h a1 = load_a16x32(h_in, H_, m0, kbase + (ki + 1) * 32, lane);
    v16h b1[4];
#pragma unroll
    for (int g = 0; g < 4; ++g)
      b1[g] = load_b32x16(WhT, H_, (g << 10) + n0, kbase + (ki + 1) * 32, lane);
    SCHED_FENCE();
#pragma unroll
    for (int g = 0; g < 4; ++g)
      acc[g] = __builtin_amdgcn_wmma_f32_16x16x32_f16(
          false, a0, false, b0[g], (short)0, acc[g], false, false);
    SCHED_FENCE();

    if (ki + 2 < 8) {
      a0 = load_a16x32(h_in, H_, m0, kbase + (ki + 2) * 32, lane);
#pragma unroll
      for (int g = 0; g < 4; ++g)
        b0[g] = load_b32x16(WhT, H_, (g << 10) + n0, kbase + (ki + 2) * 32, lane);
    }
    SCHED_FENCE();
#pragma unroll
    for (int g = 0; g < 4; ++g)
      acc[g] = __builtin_amdgcn_wmma_f32_16x16x32_f16(
          false, a1, false, b1[g], (short)0, acc[g], false, false);
    SCHED_FENCE();
  }

  // stash partials in LDS (2 x ds_store_b128 per gate)
#pragma unroll
  for (int g = 0; g < 4; ++g)
    *(v8f*)&red[w][g][lane * 8] = acc[g];
  __syncthreads();

  if (kq == 0) {                       // waves 0 and 4: reduce + pointwise
    const int wb = nt_local << 2;      // first wave of this tile's group
    v8f gsum[4];
#pragma unroll
    for (int g = 0; g < 4; ++g) {
      v8f s0 = *(const v8f*)&red[wb + 0][g][lane * 8];
      v8f s1 = *(const v8f*)&red[wb + 1][g][lane * 8];
      v8f s2 = *(const v8f*)&red[wb + 2][g][lane * 8];
      v8f s3 = *(const v8f*)&red[wb + 3][g][lane * 8];
      gsum[g] = (s0 + s1) + (s2 + s3);
    }
    const int hi8 = (lane >> 4) << 3;
    const int N = n0 + (lane & 15);
#pragma unroll
    for (int e = 0; e < 8; ++e) {
      const int M = m0 + hi8 + e;
      const size_t grow = (size_t)M * G4H;
      float gf = gsum[0][e] + xg_t[grow + N];
      float gi = gsum[1][e] + xg_t[grow + 1024 + N];
      float gg = gsum[2][e] + xg_t[grow + 2048 + N];
      float go = gsum[3][e] + xg_t[grow + 3072 + N];
      float f  = sigm(gf);
      float i  = sigm(gi);
      float gt = tanhf(gg);
      float o  = sigm(go);
      const size_t idx = (size_t)M * H_ + N;
      float cn = f * c[idx] + i * gt;
      float hn = tanhf(cn) * o;
      c[idx]   = cn;
      h32[idx] = hn;
      h_out[idx] = (_Float16)hn;
    }
  }
}

__global__ __launch_bounds__(256) void final_copy(const float* __restrict__ h32,
                                                  const float* __restrict__ c,
                                                  float* __restrict__ out) {
  int idx = blockIdx.x * blockDim.x + threadIdx.x;
  if (idx >= B_ * H_) return;
  out[idx] = h32[idx];
  out[B_ * H_ + idx] = c[idx];
}

// ---- host launcher ----------------------------------------------------------

extern "C" void kernel_launch(void* const* d_in, const int* in_sizes, int n_in,
                              void* d_out, int out_size, void* d_ws, size_t ws_size,
                              hipStream_t stream) {
  const float* x    = (const float*)d_in[0];
  const float* W_ii = (const float*)d_in[1];
  const float* W_hi = (const float*)d_in[2];
  const float* W_if = (const float*)d_in[3];
  const float* W_hf = (const float*)d_in[4];
  const float* W_ig = (const float*)d_in[5];
  const float* W_hg = (const float*)d_in[6];
  const float* W_io = (const float*)d_in[7];
  const float* W_ho = (const float*)d_in[8];
  const float* b_ii = (const float*)d_in[9];
  const float* b_hi = (const float*)d_in[10];
  const float* b_if = (const float*)d_in[11];
  const float* b_hf = (const float*)d_in[12];
  const float* b_ig = (const float*)d_in[13];
  const float* b_hg = (const float*)d_in[14];
  const float* b_io = (const float*)d_in[15];
  const float* b_ho = (const float*)d_in[16];

  // workspace layout (~593 MiB total)
  char* base = (char*)d_ws;
  size_t off = 0;
  float*     xg   = (float*)(base + off);     off += (size_t)T_ * B_ * G4H * 4;  // 512 MB
  _Float16*  xf   = (_Float16*)(base + off);  off += (size_t)B_ * T_ * I_ * 2;   // 64 MB
  _Float16*  WxT  = (_Float16*)(base + off);  off += (size_t)G4H * I_ * 2;       // 8 MB
  _Float16*  WhT  = (_Float16*)(base + off);  off += (size_t)G4H * H_ * 2;       // 8 MB
  float*     bsum = (float*)(base + off);     off += (size_t)G4H * 4;
  _Float16*  hf16 = (_Float16*)(base + off);  off += (size_t)2 * B_ * H_ * 2;    // double-buffered
  float*     h32  = (float*)(base + off);     off += (size_t)B_ * H_ * 4;
  float*     cst  = (float*)(base + off);     off += (size_t)B_ * H_ * 4;
  (void)ws_size; (void)in_sizes; (void)n_in; (void)out_size;

  // --- preprocessing
  cvt_x_f16<<<4096, 256, 0, stream>>>(x, xf, B_ * T_ * I_);
  transpose_concat<<<4096, 256, 0, stream>>>(WxT, W_if, W_ii, W_ig, W_io);
  transpose_concat<<<4096, 256, 0, stream>>>(WhT, W_hf, W_hi, W_hg, W_ho);
  build_bsum<<<16, 256, 0, stream>>>(bsum, b_if, b_hf, b_ii, b_hi,
                                     b_ig, b_hg, b_io, b_ho);
  init_state<<<64, 256, 0, stream>>>(cst, h32, hf16);

  // --- phase 1: input projection (WMMA GEMM), xg in [t][b][4H] order
  xg_gemm<<<16384, 256, 0, stream>>>(xf, WxT, bsum, xg);

  // --- phase 2: sequential recurrence; h f16 double-buffered across launches
  for (int t = 0; t < T_; ++t) {
    const _Float16* hin  = hf16 + (size_t)(t & 1) * B_ * H_;
    _Float16*       hout = hf16 + (size_t)((t + 1) & 1) * B_ * H_;
    lstm_step<<<128, 256, 0, stream>>>(xg + (size_t)t * B_ * G4H,
                                       hin, WhT, hout, cst, h32);
  }

  // --- output: (h_T, c_T) concatenated, f32
  final_copy<<<256, 256, 0, stream>>>(h32, cst, (float*)d_out);
}
